// FNO1dClassifier_90537910600153
// MI455X (gfx1250) — matrix operand order repaired and checked by reading, not compiled, not run on hardware
//
#include <hip/hip_runtime.h>
#include <hip/hip_bf16.h>
#include <math.h>

#define L_LEN 8192
#define NBATCH 64
#define CWID 32
#define NMODES 8
#define TWO_PI_F 6.28318530717958647692f
#define FSCALE 1024.0f
#define FS_INV (1.0f / 1024.0f)

typedef __attribute__((ext_vector_type(16))) _Float16 v16h;
typedef __attribute__((ext_vector_type(8)))  _Float16 v8h;
typedef __attribute__((ext_vector_type(8)))  float    v8f;
typedef __attribute__((ext_vector_type(4)))  float    v4f;

// ---------------------------------------------------------------------------
// Kernel 1: X[b,d] = sum_l x[b,l,d] * K(l),  K(l) = sum_{j=0..7} e^{-2pi i j l / L}
// Output: Xws[b*6 + 2*d + {0:re,1:im}]
// ---------------------------------------------------------------------------
__global__ __launch_bounds__(256) void k1_front_reduce(const float* __restrict__ x,
                                                       float* __restrict__ Xws) {
  __shared__ float red[256][6];
  const int b = blockIdx.x;
  const int tid = threadIdx.x;
  float ar0 = 0.f, ai0 = 0.f, ar1 = 0.f, ai1 = 0.f, ar2 = 0.f, ai2 = 0.f;
  for (int l = tid; l < L_LEN; l += 256) {
    float w = TWO_PI_F * (float)l * (1.0f / (float)L_LEN);
    float sw, cw;
    __sincosf(w, &sw, &cw);
    float Kr = 0.f, Ki = 0.f, cj = 1.f, sj = 0.f;
#pragma unroll
    for (int j = 0; j < NMODES; ++j) {
      Kr += cj; Ki -= sj;                  // e^{-i j w} = cos(jw) - i sin(jw)
      float cn = cj * cw - sj * sw;
      float sn = sj * cw + cj * sw;
      cj = cn; sj = sn;
    }
    const float* xp = x + ((size_t)b * L_LEN + l) * 3;
    float x0 = xp[0], x1 = xp[1], x2 = xp[2];
    ar0 += x0 * Kr; ai0 += x0 * Ki;
    ar1 += x1 * Kr; ai1 += x1 * Ki;
    ar2 += x2 * Kr; ai2 += x2 * Ki;
  }
  red[tid][0] = ar0; red[tid][1] = ai0; red[tid][2] = ar1;
  red[tid][3] = ai1; red[tid][4] = ar2; red[tid][5] = ai2;
  __syncthreads();
  for (int s = 128; s > 0; s >>= 1) {
    if (tid < s) {
#pragma unroll
      for (int q = 0; q < 6; ++q) red[tid][q] += red[tid + s][q];
    }
    __syncthreads();
  }
  if (tid < 6) Xws[b * 6 + tid] = red[0][tid];
}

// ---------------------------------------------------------------------------
// Kernel 2: per-batch spectral chain + fold into per-batch f16 fc1 matrix.
// Emits W1bT[b][n(64)][k(32)] (f16), n-major rows for direct B-fragment loads.
// ---------------------------------------------------------------------------
__device__ __forceinline__ void k2_stage(const float* __restrict__ wr,
                                         const float* __restrict__ wi,
                                         float* s_re, float* s_im,
                                         float (*o_re)[NMODES], float (*o_im)[NMODES],
                                         int tid, bool update_s) {
  const int o = tid >> 3, j = tid & 7;
  float ar = 0.f, ai = 0.f;
#pragma unroll 4
  for (int c = 0; c < CWID; ++c) {
    const float wrr = wr[(c * CWID + o) * NMODES + j];
    const float wii = wi[(c * CWID + o) * NMODES + j];
    const float sr = s_re[c], si = s_im[c];
    ar += sr * wrr - si * wii;
    ai += sr * wii + si * wrr;
  }
  __syncthreads();
  o_re[o][j] = ar; o_im[o][j] = ai;
  __syncthreads();
  if (update_s && tid < CWID) {
    float r = o_re[tid][0], im = 0.f;   // Im of DC is dropped by irfft
#pragma unroll
    for (int j2 = 1; j2 < NMODES; ++j2) { r += o_re[tid][j2]; im += o_im[tid][j2]; }
    s_re[tid] = r; s_im[tid] = im;
  }
  __syncthreads();
}

__global__ __launch_bounds__(256) void k2_chain(
    const float* __restrict__ Xws,
    const float* __restrict__ fc0_w, const float* __restrict__ fc0_b,
    const float* __restrict__ w1r, const float* __restrict__ w1i,
    const float* __restrict__ w2r, const float* __restrict__ w2i,
    const float* __restrict__ w3r, const float* __restrict__ w3i,
    const float* __restrict__ w4r, const float* __restrict__ w4i,
    const float* __restrict__ cw, const float* __restrict__ cb,
    const float* __restrict__ fc1_w, const float* __restrict__ fc1_b,
    _Float16* __restrict__ W1bT) {
  __shared__ float s_re[CWID], s_im[CWID];
  __shared__ float o_re[CWID][NMODES], o_im[CWID][NMODES];
  __shared__ float M[19][CWID];
  __shared__ float W32[19][64];
  const int b = blockIdx.x;
  const int tid = threadIdx.x;

  // s1[c] = sum_d X[b,d]*fc0_w[d,c] + L*fc0_b[c]
  if (tid < CWID) {
    float sr = (float)L_LEN * fc0_b[tid], si = 0.f;
#pragma unroll
    for (int d = 0; d < 3; ++d) {
      float xr = Xws[b * 6 + 2 * d + 0];
      float xi = Xws[b * 6 + 2 * d + 1];
      float w = fc0_w[d * CWID + tid];
      sr += xr * w; si += xi * w;
    }
    s_re[tid] = sr; s_im[tid] = si;
  }
  __syncthreads();

  k2_stage(w1r, w1i, s_re, s_im, o_re, o_im, tid, true);
  k2_stage(w2r, w2i, s_re, s_im, o_re, o_im, tid, true);
  k2_stage(w3r, w3i, s_re, s_im, o_re, o_im, tid, true);
  k2_stage(w4r, w4i, s_re, s_im, o_re, o_im, tid, false);  // keep out4 in o_re/o_im

  // Build folded matrix M[19][32]:
  //   rows 0..2 : A[d][o] = sum_i fc0_w[d][i]*cw[i][o]      (skip path)
  //   row  3    : sum_i fc0_b[i]*cw[i][o] + cb[o]           (skip bias)
  //   rows 4..10: FSCALE * (2/L) * Re(out4[o][1..7])        (cos coeffs)
  //   rows11..17: -FSCALE * (2/L) * Im(out4[o][1..7])       (sin coeffs)
  //   row 18    : FSCALE * Re(out4[o][0]) / L               (DC coeff)
  if (tid < 96) {
    const int d = tid >> 5, o = tid & 31;
    float a = 0.f;
    for (int i = 0; i < CWID; ++i) a += fc0_w[d * CWID + i] * cw[i * CWID + o];
    M[d][o] = a;
  } else if (tid < 128) {
    const int o = tid & 31;
    float a = cb[o];
    for (int i = 0; i < CWID; ++i) a += fc0_b[i] * cw[i * CWID + o];
    M[3][o] = a;
  } else if (tid < 160) {
    const int o = tid & 31;
    M[18][o] = FSCALE * o_re[o][0] * (1.0f / (float)L_LEN);
  }
  const float trig_s = FSCALE * 2.0f * (1.0f / (float)L_LEN);
  for (int idx = tid; idx < 14 * CWID; idx += 256) {
    const int jj = idx >> 5, o = idx & 31;
    if (jj < 7) M[4 + jj][o] = trig_s * o_re[o][jj + 1];
    else        M[11 + (jj - 7)][o] = -trig_s * o_im[o][jj - 7 + 1];
  }
  __syncthreads();

  // W32[k][n] = M[k][:] @ fc1_w[:,n]  (+ fc1_b folded into constant row 3)
  for (int idx = tid; idx < 19 * 64; idx += 256) {
    const int k = idx >> 6, n = idx & 63;
    float a = (k == 3) ? fc1_b[n] : 0.f;
#pragma unroll 4
    for (int o = 0; o < CWID; ++o) a += M[k][o] * fc1_w[o * 64 + n];
    W32[k][n] = a;
  }
  __syncthreads();

  // Emit f16, transposed [n][k], with error-compensation rows:
  //   19..21 = rows 0..2 again (paired with x_lo feature)
  //   22..24 = FSCALE * (row_d - f16(row_d))  (paired with x_hi/FSCALE feature)
  //   25     = FSCALE * (row_3 - f16(row_3))  (paired with 1/FSCALE feature)
  for (int idx = tid; idx < 64 * 32; idx += 256) {
    const int n = idx >> 5, k = idx & 31;
    float v;
    if (k < 19)       v = W32[k][n];
    else if (k < 22)  v = W32[k - 19][n];
    else if (k < 25) { float w0 = W32[k - 22][n]; v = FSCALE * (w0 - (float)(_Float16)w0); }
    else if (k == 25) { float w0 = W32[3][n];      v = FSCALE * (w0 - (float)(_Float16)w0); }
    else              v = 0.f;
    W1bT[((size_t)b * 64 + n) * 32 + k] = (_Float16)v;
  }
}

// ---------------------------------------------------------------------------
// Kernel 3: main WMMA GEMM over all (b,l) points + ReLU + fc2 epilogue.
// grid (L/128, B), 256 threads = 8 waves; each wave: one 16-row x 64-col tile.
// ---------------------------------------------------------------------------
__global__ __launch_bounds__(256) void k3_main(
    const float* __restrict__ x,
    const _Float16* __restrict__ W1bT,
    const float* __restrict__ fc2_w, const float* __restrict__ fc2_b,
    float* __restrict__ out) {
  constexpr int TS = 68;  // padded LDS row stride (floats) to dodge bank conflicts
  __shared__ float tileS[8][16 * TS];
  __shared__ float fc2wS[192];
  __shared__ float fc2bS[3];

  const int b = blockIdx.y;
  const int tid = threadIdx.x;
  const int wave = tid >> 5, lane = tid & 31;
  const int mrow = lane & 15, half = lane >> 4;

  if (tid < 192) fc2wS[tid] = fc2_w[tid];
  if (tid >= 192 && tid < 195) fc2bS[tid - 192] = fc2_b[tid - 192];
  __syncthreads();

  const int l = blockIdx.x * 128 + wave * 16 + mrow;

  // ---- build per-lane features ----
  const float* xp = x + ((size_t)b * L_LEN + l) * 3;
  const float x0 = xp[0], x1 = xp[1], x2 = xp[2];
  const float xh0 = (float)(_Float16)x0, xl0 = x0 - xh0;
  const float xh1 = (float)(_Float16)x1, xl1 = x1 - xh1;
  const float xh2 = (float)(_Float16)x2, xl2 = x2 - xh2;
  const float ang = TWO_PI_F * (float)l * (1.0f / (float)L_LEN);
  float sb, cb1;
  __sincosf(ang, &sb, &cb1);
  float C[8], S[8];
  C[1] = cb1; S[1] = sb;
#pragma unroll
  for (int j = 2; j <= 7; ++j) {
    const float cn = C[j - 1] * cb1 - S[j - 1] * sb;
    const float sn = S[j - 1] * cb1 + C[j - 1] * sb;
    C[j] = cn; S[j] = sn;
  }

  // A fragment, 16-bit A 16x32 layout:
  // lane<16 : elems 0..7 -> K0..7,  elems 8..15 -> K16..23
  // lane>=16: elems 0..7 -> K8..15, elems 8..15 -> K24..31
  v16h a;
  if (half == 0) {
    a[0] = (_Float16)x0; a[1] = (_Float16)x1; a[2] = (_Float16)x2; a[3] = (_Float16)1.0f;
    a[4] = (_Float16)(C[1] * FS_INV); a[5] = (_Float16)(C[2] * FS_INV);
    a[6] = (_Float16)(C[3] * FS_INV); a[7] = (_Float16)(C[4] * FS_INV);
    a[8] = (_Float16)(S[6] * FS_INV); a[9] = (_Float16)(S[7] * FS_INV);
    a[10] = (_Float16)FS_INV;
    a[11] = (_Float16)xl0; a[12] = (_Float16)xl1; a[13] = (_Float16)xl2;
    a[14] = (_Float16)(xh0 * FS_INV); a[15] = (_Float16)(xh1 * FS_INV);
  } else {
    a[0] = (_Float16)(C[5] * FS_INV); a[1] = (_Float16)(C[6] * FS_INV);
    a[2] = (_Float16)(C[7] * FS_INV);
    a[3] = (_Float16)(S[1] * FS_INV); a[4] = (_Float16)(S[2] * FS_INV);
    a[5] = (_Float16)(S[3] * FS_INV); a[6] = (_Float16)(S[4] * FS_INV);
    a[7] = (_Float16)(S[5] * FS_INV);
    a[8] = (_Float16)(xh2 * FS_INV); a[9] = (_Float16)FS_INV;
    a[10] = (_Float16)0.f; a[11] = (_Float16)0.f; a[12] = (_Float16)0.f;
    a[13] = (_Float16)0.f; a[14] = (_Float16)0.f; a[15] = (_Float16)0.f;
  }

  // B fragments: lanes 0-15 hold K0..7,K16..23 of column n=lane; lanes 16-31 hold K8..15,K24..31.
  const _Float16* wp = W1bT + (size_t)b * 64 * 32;
  v16h bf[4];
#pragma unroll
  for (int nt = 0; nt < 4; ++nt) {
    const int n = nt * 16 + mrow;
    const v8h lo = *(const v8h*)(wp + n * 32 + half * 8);
    const v8h hi = *(const v8h*)(wp + n * 32 + 16 + half * 8);
    bf[nt] = __builtin_shufflevector(lo, hi, 0, 1, 2, 3, 4, 5, 6, 7, 8, 9, 10, 11, 12, 13, 14, 15);
  }

  v8f acc[4];
#pragma unroll
  for (int nt = 0; nt < 4; ++nt) {
    v8f z = {};
    acc[nt] = __builtin_amdgcn_wmma_f32_16x16x32_f16(
        false, a, false, bf[nt], (short)0, z, false, false);
  }

  // ---- ReLU + transpose through LDS (per-wave private region, no barrier) ----
  float* tw = &tileS[wave][0];
#pragma unroll
  for (int nt = 0; nt < 4; ++nt) {
#pragma unroll
    for (int r = 0; r < 8; ++r) {
      const int m = r + 8 * half;           // C/D layout: lanes 16-31 hold M=r+8
      const int n = mrow + nt * 16;
      const float v = acc[nt][r];
      tw[m * TS + n] = v > 0.f ? v : 0.f;
    }
  }

  // ---- fc2: each lane reduces half the N range of one row, then xor-shuffle ----
  float p0 = 0.f, p1 = 0.f, p2 = 0.f;
  const float* trow = tw + mrow * TS + half * 32;
#pragma unroll
  for (int q = 0; q < 8; ++q) {
    const v4f v = *(const v4f*)(trow + q * 4);
#pragma unroll
    for (int e = 0; e < 4; ++e) {
      const int n = half * 32 + q * 4 + e;
      const float vv = v[e];
      p0 += vv * fc2wS[n * 3 + 0];
      p1 += vv * fc2wS[n * 3 + 1];
      p2 += vv * fc2wS[n * 3 + 2];
    }
  }
  p0 += __shfl_xor(p0, 16, 32);
  p1 += __shfl_xor(p1, 16, 32);
  p2 += __shfl_xor(p2, 16, 32);
  if (half == 0) {
    float* op = out + ((size_t)b * L_LEN + l) * 3;
    op[0] = p0 + fc2bS[0];
    op[1] = p1 + fc2bS[1];
    op[2] = p2 + fc2bS[2];
  }
}

// ---------------------------------------------------------------------------
extern "C" void kernel_launch(void* const* d_in, const int* in_sizes, int n_in,
                              void* d_out, int out_size, void* d_ws, size_t ws_size,
                              hipStream_t stream) {
  (void)in_sizes; (void)n_in; (void)out_size; (void)ws_size;
  const float* x     = (const float*)d_in[0];
  const float* fc0_w = (const float*)d_in[1];
  const float* fc0_b = (const float*)d_in[2];
  const float* cw    = (const float*)d_in[3];
  const float* cb    = (const float*)d_in[4];
  const float* fc1_w = (const float*)d_in[5];
  const float* fc1_b = (const float*)d_in[6];
  const float* fc2_w = (const float*)d_in[7];
  const float* fc2_b = (const float*)d_in[8];
  const float* w1r   = (const float*)d_in[9];
  const float* w1i   = (const float*)d_in[10];
  const float* w2r   = (const float*)d_in[11];
  const float* w2i   = (const float*)d_in[12];
  const float* w3r   = (const float*)d_in[13];
  const float* w3i   = (const float*)d_in[14];
  const float* w4r   = (const float*)d_in[15];
  const float* w4i   = (const float*)d_in[16];

  _Float16* W1bT = (_Float16*)d_ws;                                  // 64*64*32 f16 = 256 KB
  float* Xws = (float*)((char*)d_ws + (size_t)NBATCH * 64 * 32 * sizeof(_Float16));
  float* out = (float*)d_out;

  k1_front_reduce<<<dim3(NBATCH), dim3(256), 0, stream>>>(x, Xws);
  k2_chain<<<dim3(NBATCH), dim3(256), 0, stream>>>(Xws, fc0_w, fc0_b,
      w1r, w1i, w2r, w2i, w3r, w3i, w4r, w4i, cw, cb, fc1_w, fc1_b, W1bT);
  k3_main<<<dim3(L_LEN / 128, NBATCH), dim3(256), 0, stream>>>(x, W1bT, fc2_w, fc2_b, out);
}
